// Decoder_83279415870148
// MI455X (gfx1250) — matrix operand order repaired and checked by reading, not compiled, not run on hardware
//
#include <hip/hip_runtime.h>

// ---------------------------------------------------------------------------
// CDNA5 (gfx1250) transformer decoder.
//  - all matrix operands pre-converted to bf16 (weights transposed to [N][K])
//  - WMMA f32_16x16x32_bf16 for every matmul
//  - global->LDS staging via GLOBAL_LOAD_ASYNC_TO_LDS_B128 (ASYNCcnt),
//    double-buffered GEMM K-loop; straight-line staging (no divergent loops)
// ---------------------------------------------------------------------------

typedef __bf16 bf16_t;
typedef bf16_t bf16x16 __attribute__((ext_vector_type(16)));
typedef unsigned short u16x8 __attribute__((ext_vector_type(8)));
typedef float f32x8 __attribute__((ext_vector_type(8)));
typedef int i32x4 __attribute__((ext_vector_type(4)));

#define BCONST 8
#define SCONST 1024
#define DCONST 512
#define HCONST 8
#define DKCONST 64
#define DFFCONST 2048
#define LCONST 6
#define MTOT (BCONST * SCONST)

#if __has_builtin(__builtin_amdgcn_global_load_async_to_lds_b128)
#define ASYNC_LDS 1
#else
#define ASYNC_LDS 0
#endif

// round-to-nearest-even fp32 -> bf16 (raw u16)
__device__ __forceinline__ unsigned short f2bf_u(float f) {
    union { float f; unsigned u; } v; v.f = f;
    return (unsigned short)((v.u + 0x7FFFu + ((v.u >> 16) & 1u)) >> 16);
}

// 16-byte global -> LDS copy (async DMA path; ASYNCcnt tracked)
__device__ __forceinline__ void cp16(void* l, const void* g) {
#if ASYNC_LDS
    void* gnc = const_cast<void*>(g);
    __builtin_amdgcn_global_load_async_to_lds_b128(
        (__attribute__((address_space(1))) i32x4*)gnc,
        (__attribute__((address_space(3))) i32x4*)l,
        0, 0);
#else
    *(u16x8*)l = *(const u16x8*)g;
#endif
}

__device__ __forceinline__ void async_wait_all() {
#if ASYNC_LDS
#if __has_builtin(__builtin_amdgcn_s_wait_asynccnt)
    __builtin_amdgcn_s_wait_asynccnt(0);
#else
    asm volatile("s_wait_asynccnt 0x0" ::: "memory");
#endif
#endif
}

// A-fragment (16x32, MxK): lanes 0-15 row M=lane, K = {0..7,16..23};
// lanes 16-31 row M=lane-16, K = {8..15,24..31}.  (ISA 7.12.2)
__device__ __forceinline__ bf16x16 load_frag_a(const unsigned short* lds,
                                               int row, int stride, int kbase, int half) {
    union { bf16x16 v; u16x8 h[2]; } u;
    const unsigned short* p = lds + row * stride + kbase + half * 8;
    u.h[0] = *(const u16x8*)(p);
    u.h[1] = *(const u16x8*)(p + 16);
    return u.v;
}

// B-fragment (32x16, KxN) from a K-major [col][k] LDS tile.
__device__ __forceinline__ bf16x16 load_frag_b(const unsigned short* lds,
                                               int col, int stride, int kbase, int half) {
    union { bf16x16 v; u16x8 h[2]; } u;
    const unsigned short* p = lds + col * stride + kbase + half * 8;
    u.h[0] = *(const u16x8*)(p);
    u.h[1] = *(const u16x8*)(p + 16);
    return u.v;
}

__device__ __forceinline__ f32x8 wmma_bf16(bf16x16 a, bf16x16 b, f32x8 c) {
    return __builtin_amdgcn_wmma_f32_16x16x32_bf16(
        false, a, false, b, (short)0, c, false, false);
}

// ---------------------------------------------------------------------------
// One-time conversions
// ---------------------------------------------------------------------------
__global__ __launch_bounds__(256)
void f32_to_bf16_kernel(const float* __restrict__ in, unsigned short* __restrict__ out, int n) {
    int i = blockIdx.x * 256 + threadIdx.x;
    if (i < n) out[i] = f2bf_u(in[i]);
}

// W [K,N] fp32 -> Wt [N,K] bf16 (coalesced writes)
__global__ __launch_bounds__(256)
void transpose_bf16_kernel(const float* __restrict__ W, unsigned short* __restrict__ Wt,
                           int K, int N) {
    int i = blockIdx.x * 256 + threadIdx.x;
    if (i >= K * N) return;
    int n = i / K, k = i - n * K;
    Wt[(size_t)n * K + k] = f2bf_u(W[(size_t)k * N + n]);
}

// ---------------------------------------------------------------------------
// GEMM: C = act(A[M,K]b16 @ Wt[N,K]b16^T + bias), out f32 and/or bf16.
// 128 threads = 4 waves, block tile 64x64, wave tile 32x32, K step 32,
// double-buffered LDS with async staging. K is compile-time (512 or 2048)
// so all staging address math is shifts.
// ---------------------------------------------------------------------------
#define TM 64
#define TN 64
#define TK 32

template <int K>
__global__ __launch_bounds__(128)
void gemm_bf16_kernel(const unsigned short* __restrict__ A,
                      const unsigned short* __restrict__ Wt,
                      const float* __restrict__ bias,
                      float* __restrict__ Cf, unsigned short* __restrict__ Cb,
                      int N, int relu) {
    __shared__ unsigned short sA[2][TM * TK];
    __shared__ unsigned short sB[2][TN * TK];

    const int tid  = threadIdx.x;
    const int lane = tid & 31;
    const int wave = tid >> 5;
    const int half = lane >> 4;
    const int l16  = lane & 15;
    const int wm = (wave >> 1) * 32;
    const int wn = (wave & 1) * 32;
    const int tile_m = blockIdx.y * TM;
    const int tile_n = blockIdx.x * TN;

    // per-thread staging coordinates: 2 chunks per tile (rows r0 and r0+32)
    const int r0 = tid >> 2;             // 0..31
    const int c0 = (tid & 3) * 8;        // 0,8,16,24
    const unsigned short* gA0 = A  + (size_t)(tile_m + r0) * K + c0;
    const unsigned short* gB0 = Wt + (size_t)(tile_n + r0) * K + c0;
    unsigned short* lA0 = &sA[0][r0 * TK + c0];
    unsigned short* lB0 = &sB[0][r0 * TK + c0];

    auto stage = [&](int buf, int k0) {
        const size_t go = (size_t)k0;
        const int    lo = buf * (TM * TK);
        cp16(lA0 + lo,                gA0 + go);
        cp16(lA0 + lo + 32 * TK,      gA0 + go + (size_t)32 * K);
        cp16(lB0 + lo,                gB0 + go);
        cp16(lB0 + lo + 32 * TK,      gB0 + go + (size_t)32 * K);
    };

    f32x8 acc[2][2] = {};
    constexpr int nk = K / TK;
    stage(0, 0);
    int cur = 0;
    for (int kt = 0; kt < nk; ++kt) {
        async_wait_all();          // my staged tile is in LDS
        __syncthreads();           // everyone's is
        if (kt + 1 < nk) stage(cur ^ 1, (kt + 1) * TK);   // overlap DMA with math
        const unsigned short* a_ = sA[cur];
        const unsigned short* b_ = sB[cur];
        for (int im = 0; im < 2; ++im) {
            bf16x16 af = load_frag_a(a_, wm + im * 16 + l16, TK, 0, half);
            for (int in = 0; in < 2; ++in) {
                bf16x16 bf = load_frag_b(b_, wn + in * 16 + l16, TK, 0, half);
                acc[im][in] = wmma_bf16(af, bf, acc[im][in]);
            }
        }
        cur ^= 1;
    }

    for (int im = 0; im < 2; ++im) {
        for (int in = 0; in < 2; ++in) {
            const int col = tile_n + wn + in * 16 + l16;
            const float bv = bias[col];
            for (int r = 0; r < 8; ++r) {
                const int row = tile_m + wm + im * 16 + half * 8 + r;
                float v = acc[im][in][r] + bv;
                if (relu) v = fmaxf(v, 0.0f);
                if (Cf) Cf[(size_t)row * N + col] = v;
                if (Cb) Cb[(size_t)row * N + col] = f2bf_u(v);
            }
        }
    }
}

// ---------------------------------------------------------------------------
// Flash attention (causal), qkv in bf16 [M, 3D]; output bf16 [M, D].
// ---------------------------------------------------------------------------
#define QT 64
#define KT 32

__global__ __launch_bounds__(128)
void attn_kernel(const unsigned short* __restrict__ qkv, unsigned short* __restrict__ Ob) {
    constexpr int S3D = 3 * DCONST;
    const int bh    = blockIdx.x;
    const int b     = bh / HCONST;
    const int h     = bh % HCONST;
    const int qbase = blockIdx.y * QT;

    const int tid  = threadIdx.x;
    const int lane = tid & 31;
    const int wave = tid >> 5;
    const int half = lane >> 4;
    const int l16  = lane & 15;

    __shared__ unsigned short sQ[QT * DKCONST];    // [qrow][feat]
    __shared__ unsigned short sK[KT * DKCONST];    // [key][feat]
    __shared__ unsigned short sV[DKCONST * KT];    // [feat][key]
    __shared__ unsigned short sP[4][16 * KT];      // per-wave P relayout

    const size_t base = (size_t)b * SCONST * S3D;

    // per-thread staging coordinates: row rs0 (+16/+32/+48), feat chunk cs0
    const int rs0 = tid >> 3;            // 0..15
    const int cs0 = (tid & 7) * 8;       // 0..56

    // stage Q tile (4 chunks per thread, async)
    {
        const unsigned short* gQ = qkv + base + (size_t)(qbase + rs0) * S3D
                                   + h * DKCONST + cs0;
        unsigned short* lQ = &sQ[rs0 * DKCONST + cs0];
        cp16(lQ,                 gQ);
        cp16(lQ + 16 * DKCONST,  gQ + (size_t)16 * S3D);
        cp16(lQ + 32 * DKCONST,  gQ + (size_t)32 * S3D);
        cp16(lQ + 48 * DKCONST,  gQ + (size_t)48 * S3D);
    }
    async_wait_all();
    __syncthreads();

    bf16x16 qf[2];
    {
        const int row = wave * 16 + l16;
        qf[0] = load_frag_a(sQ, row, DKCONST, 0, half);
        qf[1] = load_frag_a(sQ, row, DKCONST, 32, half);
    }

    float mrow[8], lrow[8];
    for (int r = 0; r < 8; ++r) { mrow[r] = -1e30f; lrow[r] = 0.0f; }
    f32x8 oacc[4] = {};

    const int nkt = (qbase + QT) / KT;
    for (int kt = 0; kt < nkt; ++kt) {
        const int kg = kt * KT;
        // K tile [key][feat]: 2 chunks per thread, async
        {
            const unsigned short* gK = qkv + base + (size_t)(kg + rs0) * S3D
                                       + DCONST + h * DKCONST + cs0;
            unsigned short* lK = &sK[rs0 * DKCONST + cs0];
            cp16(lK,                gK);
            cp16(lK + 16 * DKCONST, gK + (size_t)16 * S3D);
        }
        // V tile transposed to [feat][key]: vector load + 16-bit scatter
        {
            const unsigned short* gV = qkv + base + (size_t)(kg + rs0) * S3D
                                       + 2 * DCONST + h * DKCONST + cs0;
            u16x8 v0 = *(const u16x8*)(gV);
            u16x8 v1 = *(const u16x8*)(gV + (size_t)16 * S3D);
            for (int j = 0; j < 8; ++j) {
                sV[(cs0 + j) * KT + rs0]      = v0[j];
                sV[(cs0 + j) * KT + rs0 + 16] = v1[j];
            }
        }
        async_wait_all();
        __syncthreads();

        // scores: 16q x 32k (two 16x16 tiles), DK=64 = 2 K-chunks
        f32x8 sc[2] = {};
        for (int sub = 0; sub < 2; ++sub) {
            bf16x16 bk0 = load_frag_b(sK, sub * 16 + l16, DKCONST, 0, half);
            bf16x16 bk1 = load_frag_b(sK, sub * 16 + l16, DKCONST, 32, half);
            sc[sub] = wmma_bf16(qf[0], bk0, sc[sub]);
            sc[sub] = wmma_bf16(qf[1], bk1, sc[sub]);
        }

        // online softmax: row M = r + 8*half across the 16 lanes of this half
        const int qrb = qbase + wave * 16;
        float pexp[2][8];
        for (int r = 0; r < 8; ++r) {
            const int qr = qrb + half * 8 + r;
            float s0 = sc[0][r] * 0.125f;
            float s1 = sc[1][r] * 0.125f;
            if (kg + l16 > qr)      s0 = -1e30f;
            if (kg + 16 + l16 > qr) s1 = -1e30f;
            float mx = fmaxf(s0, s1);
            for (int off = 8; off >= 1; off >>= 1)
                mx = fmaxf(mx, __shfl_xor(mx, off, 32));
            mx = fmaxf(mx, mrow[r]);
            const float e0 = __expf(s0 - mx);
            const float e1 = __expf(s1 - mx);
            float rs = e0 + e1;
            for (int off = 8; off >= 1; off >>= 1)
                rs += __shfl_xor(rs, off, 32);
            const float corr = __expf(mrow[r] - mx);
            lrow[r] = lrow[r] * corr + rs;
            mrow[r] = mx;
            pexp[0][r] = e0; pexp[1][r] = e1;
            for (int t = 0; t < 4; ++t) oacc[t][r] *= corr;
        }

        // P: C-layout -> A-layout via wave-private LDS patch (DS in-order per wave)
        unsigned short* myP = sP[wave];
        for (int sub = 0; sub < 2; ++sub)
            for (int r = 0; r < 8; ++r)
                myP[(half * 8 + r) * KT + sub * 16 + l16] = f2bf_u(pexp[sub][r]);
        asm volatile("" ::: "memory");

        // O += P @ V
        bf16x16 pa = load_frag_a(myP, l16, KT, 0, half);
        for (int t = 0; t < 4; ++t) {
            bf16x16 vb = load_frag_b(sV, t * 16 + l16, KT, 0, half);
            oacc[t] = wmma_bf16(pa, vb, oacc[t]);
        }
        __syncthreads();
    }

    for (int t = 0; t < 4; ++t) {
        const int feat = h * DKCONST + t * 16 + l16;
        for (int r = 0; r < 8; ++r) {
            const int qr = qbase + wave * 16 + half * 8 + r;
            Ob[((size_t)b * SCONST + qr) * DCONST + feat] = f2bf_u(oacc[t][r] / lrow[r]);
        }
    }
}

// ---------------------------------------------------------------------------
// outF = LayerNorm(a + badd)*g + beta over D=512; also bf16 copy if requested.
// ---------------------------------------------------------------------------
__global__ __launch_bounds__(256)
void add_ln_kernel(const float* __restrict__ a, const float* __restrict__ badd,
                   const float* __restrict__ g, const float* __restrict__ beta,
                   float* __restrict__ outF, unsigned short* __restrict__ outB) {
    const int row = blockIdx.x;
    const int tid = threadIdx.x;
    const size_t off = (size_t)row * DCONST;

    float v0 = a[off + tid] + badd[off + tid];
    float v1 = a[off + tid + 256] + badd[off + tid + 256];

    float s  = v0 + v1;
    float sq = v0 * v0 + v1 * v1;
    for (int o = 16; o >= 1; o >>= 1) {
        s  += __shfl_xor(s,  o, 32);
        sq += __shfl_xor(sq, o, 32);
    }
    __shared__ float rs[8], rq[8];
    if ((tid & 31) == 0) { rs[tid >> 5] = s; rq[tid >> 5] = sq; }
    __syncthreads();
    float ts = 0.f, tq = 0.f;
    for (int i = 0; i < 8; ++i) { ts += rs[i]; tq += rq[i]; }

    const float mean = ts * (1.0f / DCONST);
    const float var  = tq * (1.0f / DCONST) - mean * mean;
    const float rstd = rsqrtf(var + 1e-5f);

    const float o0 = (v0 - mean) * rstd * g[tid]       + beta[tid];
    const float o1 = (v1 - mean) * rstd * g[tid + 256] + beta[tid + 256];
    if (outF) { outF[off + tid] = o0; outF[off + tid + 256] = o1; }
    if (outB) { outB[off + tid] = f2bf_u(o0); outB[off + tid + 256] = f2bf_u(o1); }
}

// ---------------------------------------------------------------------------
// Orchestration
// ---------------------------------------------------------------------------
extern "C" void kernel_launch(void* const* d_in, const int* in_sizes, int n_in,
                              void* d_out, int out_size, void* d_ws, size_t ws_size,
                              hipStream_t stream) {
    (void)in_sizes; (void)n_in; (void)out_size; (void)ws_size;

    const float* x      = (const float*)d_in[0];
    const float* qkv_w  = (const float*)d_in[1];
    const float* qkv_b  = (const float*)d_in[2];
    const float* out_w  = (const float*)d_in[3];
    const float* out_b  = (const float*)d_in[4];
    const float* ln1_g  = (const float*)d_in[5];
    const float* ln1_b  = (const float*)d_in[6];
    const float* mlp_w1 = (const float*)d_in[7];
    const float* mlp_b1 = (const float*)d_in[8];
    const float* mlp_w2 = (const float*)d_in[9];
    const float* mlp_b2 = (const float*)d_in[10];
    const float* ln2_g  = (const float*)d_in[11];
    const float* ln2_b  = (const float*)d_in[12];

    // workspace layout
    float* xbuf = (float*)d_ws;                                  // [M,D] f32
    float* tbuf = xbuf + (size_t)MTOT * DCONST;                  // [M,D] f32
    unsigned short* xb   = (unsigned short*)(tbuf + (size_t)MTOT * DCONST); // [M,D]
    unsigned short* qkvb = xb + (size_t)MTOT * DCONST;           // [M,3D]
    unsigned short* attb = qkvb + (size_t)MTOT * 3 * DCONST;     // [M,D]
    unsigned short* hb   = attb + (size_t)MTOT * DCONST;         // [M,DFF]
    unsigned short* wq   = hb + (size_t)MTOT * DFFCONST;         // [L][3D][D]
    unsigned short* wo   = wq + (size_t)LCONST * 3 * DCONST * DCONST;   // [L][D][D]
    unsigned short* w1   = wo + (size_t)LCONST * DCONST * DCONST;       // [L][DFF][D]
    unsigned short* w2   = w1 + (size_t)LCONST * DFFCONST * DCONST;     // [L][D][DFF]

    // one-time: x -> bf16, weights -> bf16 transposed [N][K]
    f32_to_bf16_kernel<<<dim3(MTOT * DCONST / 256), dim3(256), 0, stream>>>(
        x, xb, MTOT * DCONST);
    for (int l = 0; l < LCONST; ++l) {
        transpose_bf16_kernel<<<dim3(DCONST * 3 * DCONST / 256), dim3(256), 0, stream>>>(
            qkv_w + (size_t)l * DCONST * 3 * DCONST,
            wq + (size_t)l * 3 * DCONST * DCONST, DCONST, 3 * DCONST);
        transpose_bf16_kernel<<<dim3(DCONST * DCONST / 256), dim3(256), 0, stream>>>(
            out_w + (size_t)l * DCONST * DCONST,
            wo + (size_t)l * DCONST * DCONST, DCONST, DCONST);
        transpose_bf16_kernel<<<dim3(DCONST * DFFCONST / 256), dim3(256), 0, stream>>>(
            mlp_w1 + (size_t)l * DCONST * DFFCONST,
            w1 + (size_t)l * DFFCONST * DCONST, DCONST, DFFCONST);
        transpose_bf16_kernel<<<dim3(DFFCONST * DCONST / 256), dim3(256), 0, stream>>>(
            mlp_w2 + (size_t)l * DFFCONST * DCONST,
            w2 + (size_t)l * DCONST * DFFCONST, DFFCONST, DCONST);
    }

    const dim3 blk(128);
    const unsigned short* curB = xb;   // bf16 view of current x
    const float* curF = x;             // f32 view of current x (residual)
    for (int l = 0; l < LCONST; ++l) {
        // qkv (bf16 out only), K = D
        gemm_bf16_kernel<DCONST><<<dim3(3 * DCONST / TN, MTOT / TM), blk, 0, stream>>>(
            curB, wq + (size_t)l * 3 * DCONST * DCONST,
            qkv_b + (size_t)l * 3 * DCONST, nullptr, qkvb, 3 * DCONST, 0);
        // causal flash attention (bf16 in/out)
        attn_kernel<<<dim3(BCONST * HCONST, SCONST / QT), blk, 0, stream>>>(qkvb, attb);
        // out-proj (f32 out only), K = D
        gemm_bf16_kernel<DCONST><<<dim3(DCONST / TN, MTOT / TM), blk, 0, stream>>>(
            attb, wo + (size_t)l * DCONST * DCONST,
            out_b + (size_t)l * DCONST, tbuf, nullptr, DCONST, 0);
        // x = LN1(proj + x)  -> f32 + bf16
        add_ln_kernel<<<dim3(MTOT), dim3(256), 0, stream>>>(
            tbuf, curF, ln1_g + (size_t)l * DCONST, ln1_b + (size_t)l * DCONST,
            xbuf, xb);
        // h = relu(x @ w1 + b1)  (bf16 out only), K = D
        gemm_bf16_kernel<DCONST><<<dim3(DFFCONST / TN, MTOT / TM), blk, 0, stream>>>(
            xb, w1 + (size_t)l * DFFCONST * DCONST,
            mlp_b1 + (size_t)l * DFFCONST, nullptr, hb, DFFCONST, 1);
        // t = h @ w2 + b2  (f32 out only), K = DFF
        gemm_bf16_kernel<DFFCONST><<<dim3(DCONST / TN, MTOT / TM), blk, 0, stream>>>(
            hb, w2 + (size_t)l * DCONST * DFFCONST,
            mlp_b2 + (size_t)l * DCONST, tbuf, nullptr, DCONST, 0);
        // x = LN2(t + x): final layer -> d_out (f32), else xbuf + xb
        float* dstF = (l == LCONST - 1) ? (float*)d_out : xbuf;
        unsigned short* dstB = (l == LCONST - 1) ? nullptr : xb;
        add_ln_kernel<<<dim3(MTOT), dim3(256), 0, stream>>>(
            tbuf, xbuf, ln2_g + (size_t)l * DCONST, ln2_b + (size_t)l * DCONST,
            dstF, dstB);
        curB = xb;
        curF = xbuf;
    }
}
